// HCHAlayer_14499809591686
// MI455X (gfx1250) — matrix operand order repaired and checked by reading, not compiled, not run on hardware
//
#include <hip/hip_runtime.h>
#include <math.h>

// ---------------------------------------------------------------------------
// HCHA layer for MI455X (gfx1250): bf16 WMMA GEMMs with on-the-fly attention
// matrix (Hm) regeneration, fused normalization epilogues.
// LDS layouts chosen so WMMA fragments load as aligned ds_load_b128; all 8
// B fragments preloaded per K-step so DS latency overlaps the WMMA pipe.
// ---------------------------------------------------------------------------

typedef __attribute__((ext_vector_type(16))) __bf16 v16bf;
typedef __attribute__((ext_vector_type(8)))  float  v8f;

#define NEG_SLOPE 0.1f
#define LDSP 40   // LDS row pitch in bf16 elements (80 B = 16B-aligned, 20 words)

__device__ __forceinline__ unsigned short f2bf(float f) {
    unsigned int u = __float_as_uint(f);
    u += 0x7FFFu + ((u >> 16) & 1u);   // round-to-nearest-even
    return (unsigned short)(u >> 16);
}

// act8(s) = leaky_relu(8*tanh(s/8))
__device__ __forceinline__ float act8(float s) {
    float t = 8.0f * tanhf(s * 0.125f);
    return t >= 0.0f ? t : NEG_SLOPE * t;
}

enum { A_PLAIN = 0, A_HT = 1, A_HMT = 2, A_HM = 3 };
enum { EPI_NONE = 0, EPI_RSCALE = 1, EPI_LEAKY = 2 };

union Pack4 { unsigned short s[4]; uint2 u; };
union Frag  { v16bf v; uint4 q[2]; };

// Generic 128x128-tile GEMM, C(f32) = A(bf16-staged) @ B(bf16-staged).
// A source modes:
//   A_PLAIN : A[m,k] = Asrc[m*lda + k]
//   A_HT    : A[m,k] = Asrc[k*lda + m]                       (H^T for he)
//   A_HMT   : A[e,i] = exp(act8(s1[i]+s2[e]))*H[i*lda+e]*invrow[i]
//   A_HM    : A[i,e] = exp(act8(s1[i]+s2[e]))*H[i*lda+e]*invrow[i]
// Epilogues: none / divide row by rscale[row] / leaky_relu.
template <int AMODE, int EPI>
__global__ __launch_bounds__(256) void gemm_bf16_kernel(
    const float* __restrict__ A, int lda,
    const float* __restrict__ B, int ldb,
    float* __restrict__ C, int ldc,
    int M, int Nn, int K,
    const float* __restrict__ s1, const float* __restrict__ s2,
    const float* __restrict__ invrow, const float* __restrict__ rscale)
{
    // A tile: [mm 0..127][kk 0..31], pitch LDSP -> lane fragment = 2x b128
    __shared__ alignas(16) unsigned short lds_a[128 * LDSP];
    // B tile: [nn 0..127][kk 0..31], pitch LDSP (K-contiguous per column)
    __shared__ alignas(16) unsigned short lds_b[128 * LDSP];

    const int tid  = threadIdx.x;
    const int lane = tid & 31;
    const int wave = tid >> 5;
    const int m0 = blockIdx.x * 128;
    const int n0 = blockIdx.y * 128;

    v8f zero = {0.f, 0.f, 0.f, 0.f, 0.f, 0.f, 0.f, 0.f};
    v8f acc[8];
#pragma unroll
    for (int i = 0; i < 8; ++i) acc[i] = zero;

    const int mmA   = (wave << 4) + (lane & 15);
    const int koffA = (lane >> 4) << 3;   // lanes 0-15: K 0..7 / 16..23; 16-31: 8..15 / 24..31
    const int koffB = (lane >> 4) << 3;   // lanes 0-15: K 0..7 / 16..23; 16-31: 8..15 / 24..31

    for (int k0 = 0; k0 < K; k0 += 32) {
        // ---- stage A tile: 4 bf16 per thread per iter, ds_store_b64 ----
        if (AMODE == A_PLAIN || AMODE == A_HM) {
            // kk-fastest quads: coalesced float4 global loads
#pragma unroll
            for (int i = 0; i < 4; ++i) {
                int q  = tid + 256 * i;       // 0..1023
                int kk = (q & 7) << 2;
                int mm = q >> 3;
                int gm = m0 + mm;
                int gk = k0 + kk;
                float4 f = *(const float4*)&A[(size_t)gm * lda + gk];
                float v[4] = {f.x, f.y, f.z, f.w};
                if (AMODE == A_HM) {          // m = node i, k = edge e
                    float sm = s1[gm], ir = invrow[gm];
#pragma unroll
                    for (int j = 0; j < 4; ++j)
                        v[j] = __expf(act8(sm + s2[gk + j])) * v[j] * ir;
                }
                Pack4 pk;
#pragma unroll
                for (int j = 0; j < 4; ++j) pk.s[j] = f2bf(v[j]);
                *(uint2*)&lds_a[mm * LDSP + kk] = pk.u;
            }
        } else {
            // transposed source: mm-fastest (coalesced scalar streams per j)
#pragma unroll
            for (int i = 0; i < 4; ++i) {
                int q  = tid + 256 * i;
                int mm = q & 127;
                int kk = (q >> 7) << 2;
                int gm = m0 + mm;
                int gk = k0 + kk;
                Pack4 pk;
#pragma unroll
                for (int j = 0; j < 4; ++j) {
                    float h = A[(size_t)(gk + j) * lda + gm];
                    float v;
                    if (AMODE == A_HT) v = h;  // he: plain H^T
                    else               v = __expf(act8(s1[gk + j] + s2[gm])) * h * invrow[gk + j];
                    pk.s[j] = f2bf(v);
                }
                *(uint2*)&lds_a[mm * LDSP + kk] = pk.u;
            }
        }
        // ---- stage B tile into [nn][kk] (K-contiguous), nn-fastest ----
#pragma unroll
        for (int i = 0; i < 4; ++i) {
            int q  = tid + 256 * i;
            int nn = q & 127;
            int kk = (q >> 7) << 2;
            Pack4 pk;
#pragma unroll
            for (int j = 0; j < 4; ++j)
                pk.s[j] = f2bf(B[(size_t)(k0 + kk + j) * ldb + n0 + nn]);
            *(uint2*)&lds_b[nn * LDSP + kk] = pk.u;
        }
        __syncthreads();

        // ---- preload A + all 8 B fragments (16+2 ds_load_b128, pipelined) ----
        Frag af;
        af.q[0] = *(const uint4*)&lds_a[mmA * LDSP + koffA];
        af.q[1] = *(const uint4*)&lds_a[mmA * LDSP + koffA + 16];
        Frag bf[8];
#pragma unroll
        for (int nt = 0; nt < 8; ++nt) {
            int nnB = (nt << 4) + (lane & 15);
            bf[nt].q[0] = *(const uint4*)&lds_b[nnB * LDSP + koffB];
            bf[nt].q[1] = *(const uint4*)&lds_b[nnB * LDSP + koffB + 16];
        }
        // ---- 8 back-to-back WMMAs ----
#pragma unroll
        for (int nt = 0; nt < 8; ++nt)
            acc[nt] = __builtin_amdgcn_wmma_f32_16x16x32_bf16(
                false, af.v, false, bf[nt].v, (short)0, acc[nt], false, false);
        __syncthreads();
    }

    // ---- epilogue: C layout per ISA (VGPR r -> M = r + 8*(lane>=16)) ----
    const int rbase = m0 + (wave << 4) + ((lane >> 4) << 3);
    const int cbase = n0 + (lane & 15);
    float rs[8];
    if (EPI == EPI_RSCALE) {
#pragma unroll
        for (int r = 0; r < 8; ++r) rs[r] = 1.0f / rscale[rbase + r];
    }
#pragma unroll
    for (int nt = 0; nt < 8; ++nt) {
        int col = cbase + (nt << 4);
#pragma unroll
        for (int r = 0; r < 8; ++r) {
            float v = acc[nt][r];
            if (EPI == EPI_RSCALE) v *= rs[r];
            if (EPI == EPI_LEAKY)  v = (v >= 0.f) ? v : NEG_SLOPE * v;
            C[(size_t)(rbase + r) * ldc + col] = v;
        }
    }
}

// Column sums over the node dimension. HM=0: colsum of H. HM=1: colsum of Hm.
template <int HM>
__global__ __launch_bounds__(256) void colsum_kernel(
    const float* __restrict__ H, int Nrows, int E,
    const float* __restrict__ s1, const float* __restrict__ s2,
    const float* __restrict__ invrow, float* __restrict__ out)
{
    int e = blockIdx.x * blockDim.x + threadIdx.x;
    float s2e = HM ? s2[e] : 0.f;
    int chunk = (Nrows + gridDim.y - 1) / gridDim.y;
    int nbeg = blockIdx.y * chunk;
    int nend = nbeg + chunk; if (nend > Nrows) nend = Nrows;
    float acc = 0.f;
    for (int n = nbeg; n < nend; ++n) {
        float h = H[(size_t)n * E + e];
        if (HM) acc += __expf(act8(s1[n] + s2e)) * h * invrow[n];
        else    acc += h;
    }
    atomicAdd(&out[e], acc);
}

// s1[i] = dot(xp[i,:], a1); s2[e] = dot(hep[e,:], a2). One 256-thread block per row.
__global__ __launch_bounds__(256) void rowdot_kernel(
    const float* __restrict__ xp, const float* __restrict__ hep,
    const float* __restrict__ a1, const float* __restrict__ a2,
    float* __restrict__ s1, float* __restrict__ s2, int N)
{
    __shared__ float red[256];
    int row = blockIdx.x;
    const float *src, *a; float* dst; int r;
    if (row < N) { src = xp;  a = a1; dst = s1; r = row; }
    else         { src = hep; a = a2; dst = s2; r = row - N; }
    int t = threadIdx.x;
    red[t] = src[(size_t)r * 256 + t] * a[t];
    __syncthreads();
    for (int s = 128; s > 0; s >>= 1) {
        if (t < s) red[t] += red[t + s];
        __syncthreads();
    }
    if (t == 0) dst[r] = red[0];
}

// invrow[i] = 1 / sum_j exp(act8(s1[i]+s2[j])) * H[i,j]
__global__ __launch_bounds__(256) void rowsum_kernel(
    const float* __restrict__ H, const float* __restrict__ s1,
    const float* __restrict__ s2, float* __restrict__ invrow, int E)
{
    __shared__ float red[256];
    int i = blockIdx.x;
    int t = threadIdx.x;
    float s1i = s1[i];
    float acc = 0.f;
    for (int j = t; j < E; j += 256)
        acc += __expf(act8(s1i + s2[j])) * H[(size_t)i * E + j];
    red[t] = acc;
    __syncthreads();
    for (int s = 128; s > 0; s >>= 1) {
        if (t < s) red[t] += red[t + s];
        __syncthreads();
    }
    if (t == 0) invrow[i] = 1.0f / red[0];
}

extern "C" void kernel_launch(void* const* d_in, const int* in_sizes, int n_in,
                              void* d_out, int out_size, void* d_ws, size_t ws_size,
                              hipStream_t stream)
{
    const int N = 12288, E = 6144, F = 256;
    const float* x  = (const float*)d_in[0];   // [N,F]
    const float* H  = (const float*)d_in[1];   // [N,E]
    const float* p  = (const float*)d_in[2];   // [F,F]
    const float* a1 = (const float*)d_in[3];   // [F,1]
    const float* a2 = (const float*)d_in[4];   // [F,1]
    float* out = (float*)d_out;                // [N,F]

    float* ws        = (float*)d_ws;
    float* sumHT     = ws;                      // E   (atomic-accumulated)
    float* colsumHm  = sumHT + E;               // E   (atomic-accumulated)
    float* s1        = colsumHm + E;            // N
    float* s2        = s1 + N;                  // E
    float* invrow    = s2 + E;                  // N
    float* he        = invrow + N;              // E*F
    float* xp        = he + (size_t)E * F;      // N*F
    float* hep       = xp + (size_t)N * F;      // E*F
    float* tb        = hep + (size_t)E * F;     // E*F
    // total workspace ~31.6 MB

    hipMemsetAsync(sumHT, 0, sizeof(float) * 2 * E, stream);

    dim3 blk(256);

    // 1) sumHT[e] = colsum of H
    colsum_kernel<0><<<dim3(E / 256, 32), blk, 0, stream>>>(H, N, E, nullptr, nullptr, nullptr, sumHT);
    // 2) he = (H^T @ x) / sumHT        (M=E, Nn=F, K=N)
    gemm_bf16_kernel<A_HT, EPI_RSCALE><<<dim3(E / 128, F / 128), blk, 0, stream>>>(
        H, E, x, F, he, F, E, F, N, nullptr, nullptr, nullptr, sumHT);
    // 3) xp = x @ p                    (M=N, Nn=F, K=F)
    gemm_bf16_kernel<A_PLAIN, EPI_NONE><<<dim3(N / 128, F / 128), blk, 0, stream>>>(
        x, F, p, F, xp, F, N, F, F, nullptr, nullptr, nullptr, nullptr);
    // 4) hep = he @ p                  (M=E, Nn=F, K=F)
    gemm_bf16_kernel<A_PLAIN, EPI_NONE><<<dim3(E / 128, F / 128), blk, 0, stream>>>(
        he, F, p, F, hep, F, E, F, F, nullptr, nullptr, nullptr, nullptr);
    // 5) s1 = xp@a1, s2 = hep@a2
    rowdot_kernel<<<dim3(N + E), blk, 0, stream>>>(xp, hep, a1, a2, s1, s2, N);
    // 6) invrow[i] = 1 / rowsum of un-normalized Hm
    rowsum_kernel<<<dim3(N), blk, 0, stream>>>(H, s1, s2, invrow, E);
    // 7) colsumHm[e] = colsum of normalized Hm  (Binv = 1/colsumHm, applied in epilogue)
    colsum_kernel<1><<<dim3(E / 256, 32), blk, 0, stream>>>(H, N, E, s1, s2, invrow, colsumHm);
    // 8) tb = Binv * (Hm^T @ xp)       (M=E, Nn=F, K=N; D==1 so D*xp == xp)
    gemm_bf16_kernel<A_HMT, EPI_RSCALE><<<dim3(E / 128, F / 128), blk, 0, stream>>>(
        H, E, xp, F, tb, F, E, F, N, s1, s2, invrow, colsumHm);
    // 9) out = leaky_relu(Hm @ tb)     (M=N, Nn=F, K=E; D==1)
    gemm_bf16_kernel<A_HM, EPI_LEAKY><<<dim3(N / 128, F / 128), blk, 0, stream>>>(
        H, E, tb, F, out, F, N, F, E, s1, s2, invrow, nullptr);
}
